// SelfAttention_32255204393403
// MI455X (gfx1250) — matrix operand order repaired and checked by reading, not compiled, not run on hardware
//
#include <hip/hip_runtime.h>
#include <stdint.h>

// ---------------- types ----------------
typedef __attribute__((ext_vector_type(16))) _Float16 v16h;
typedef __attribute__((ext_vector_type(8)))  _Float16 v8h;
typedef __attribute__((ext_vector_type(8)))  float    v8f;

#define NTOK 4128      // 4096 visual + 32 grounding tokens
#define QD   320       // query dim == inner dim
#define NH   8
#define DHD  40        // dim head
#define DP   64        // padded dim head
#define W2D  4096      // 64*64 visual tokens

// ---------------- helpers ----------------
static __device__ __forceinline__ v8f wmma_f16(v16h a, v16h b, v8f c) {
  // D = A(16x32 f16) * B(32x16 f16) + C(16x16 f32)
  return __builtin_amdgcn_wmma_f32_16x16x32_f16(false, a, false, b, (short)0, c,
                                                false, false);
}

// A-operand layout (16x32 f16): lane L holds row (L&15);
// halves 0..7  <- K = 16*(L>>4) + 0..7
// halves 8..15 <- K = 16*(L>>4) + 8 + 0..7
static __device__ __forceinline__ v16h load_a_h(const _Float16* row, int kh) {
  v8h lo = *(const v8h*)(row + 2 * kh);
  v8h hi = *(const v8h*)(row + 2 * kh + 8);
  v16h r;
#pragma unroll
  for (int t = 0; t < 8; ++t) { r[t] = lo[t]; r[t + 8] = hi[t]; }
  return r;
}

// ---------------- prep kernels ----------------
__global__ void k_zero(uint32_t* p, long n) {
  long i = (long)blockIdx.x * blockDim.x + threadIdx.x;
  if (i < n) p[i] = 0u;
}

// x (f32) -> x16 (f16), one pass
__global__ void k_x16(const float* x, _Float16* x16, long n) {
  long i = (long)blockIdx.x * blockDim.x + threadIdx.x;
  if (i < n) x16[i] = (_Float16)x[i];
}

// transpose 4 weight matrices [k][n] f32 -> [n][k] f16 (B-operand friendly)
__global__ void k_wt(const float* wq, const float* wk, const float* wv, const float* wo,
                     _Float16* tq, _Float16* tk, _Float16* tv, _Float16* to_) {
  int t = blockIdx.x * blockDim.x + threadIdx.x;
  if (t >= QD * QD) return;
  int k = t / QD, n = t % QD;
  int d = n * QD + k;
  tq[d] = (_Float16)wq[t];
  tk[d] = (_Float16)wk[t];
  tv[d] = (_Float16)wv[t];
  to_[d] = (_Float16)wo[t];
}

// per-token mask code word.
// visual i: bits 0..7 = object membership bitmap.
// grounding r in [8,24): 0x8100 (bit15 = keep-all).
// grounding r in [0,8)u[24,32): (1<<(r&7)) | 0x100 (bit8 = grounding-common).
// keep(i,j) = (i==j) | ((ci|cj)>>15) | ((ci & cj & 0x1FF) != 0)
__global__ void k_code(const float* att, unsigned short* code) {
  int i = blockIdx.x * blockDim.x + threadIdx.x;
  if (i >= NTOK) return;
  if (i < W2D) {
    unsigned c = 0;
#pragma unroll
    for (int o = 0; o < NH; ++o)
      if (att[o * W2D + i] != 0.0f) c |= (1u << o);
    code[i] = (unsigned short)c;
  } else {
    int r = i - W2D;
    unsigned c = (r >= 8 && r < 24) ? 0x8100u : ((1u << (r & 7)) | 0x100u);
    code[i] = (unsigned short)c;
  }
}

// ---------------- QKV projection: [N,320] x [320,320] (x3) ----------------
// grid.x = NTOK/16, block = 640 (20 waves, one 16-col tile each)
__global__ void __launch_bounds__(640) k_qkv(const _Float16* x16,
                                             const _Float16* wqt, const _Float16* wkt,
                                             const _Float16* wvt,
                                             _Float16* Qp, _Float16* Kp, _Float16* Vt) {
  const int lane = threadIdx.x & 31;
  const int wid  = threadIdx.x >> 5;       // ntile 0..19
  const int hi   = lane >> 4;
  const int col  = lane & 15;
  const int kh   = 8 * hi;
  const int kb   = 16 * hi;
  const int i0   = blockIdx.x * 16;
  const int n0   = wid * 16;

  v8f aq = {}, ak = {}, av = {};
  const _Float16* xrow  = x16 + (size_t)(i0 + col) * QD;
  const _Float16* bqrow = wqt + (size_t)(n0 + col) * QD;
  const _Float16* bkrow = wkt + (size_t)(n0 + col) * QD;
  const _Float16* bvrow = wvt + (size_t)(n0 + col) * QD;

#pragma unroll
  for (int k0 = 0; k0 < QD; k0 += 32) {
    v16h xa = load_a_h(xrow + k0, kh);
    v16h bq = *(const v16h*)(bqrow + k0 + kb);
    v16h bk = *(const v16h*)(bkrow + k0 + kb);
    v16h bv = *(const v16h*)(bvrow + k0 + kb);
    aq = wmma_f16(xa, bq, aq);
    ak = wmma_f16(xa, bk, ak);
    av = wmma_f16(xa, bv, av);
  }

  const int jn = n0 + col;       // inner column
  const int h  = jn / DHD;
  const int c  = jn - h * DHD;   // 0..39
#pragma unroll
  for (int r = 0; r < 8; ++r) {
    int i = i0 + r + 8 * hi;
    Qp[((size_t)h * NTOK + i) * DP + c] = (_Float16)aq[r];
    Kp[((size_t)h * NTOK + i) * DP + c] = (_Float16)ak[r];
    Vt[((size_t)h * DP + c) * NTOK + i] = (_Float16)av[r];   // V transposed
  }
}

// ---------------- flash attention ----------------
// grid.x = NTOK/16 query tiles, block = 256 (8 waves, one head each)
__global__ void __launch_bounds__(256) k_attn(const _Float16* Qp, const _Float16* Kp,
                                              const _Float16* Vt,
                                              const unsigned short* code, _Float16* Of) {
  __shared__ __attribute__((aligned(32))) _Float16 plds[8][16][32]; // 1KB per wave
  const int lane = threadIdx.x & 31;
  const int h    = threadIdx.x >> 5;
  const int hi   = lane >> 4;
  const int col  = lane & 15;
  const int kh   = 8 * hi;
  const int kb   = 16 * hi;
  const int i0   = blockIdx.x * 16;
  const float scale = 0.15811388300841898f;   // 40^-0.5
  const float NEG = -1e30f;

  const _Float16* Qh = Qp + (size_t)h * NTOK * DP;
  const _Float16* Kh = Kp + (size_t)h * NTOK * DP;
  const _Float16* Vh = Vt + (size_t)h * DP * NTOK;

  v16h qa0 = load_a_h(Qh + (size_t)(i0 + col) * DP, kh);        // K = c 0..31
  v16h qa1 = load_a_h(Qh + (size_t)(i0 + col) * DP + 32, kh);   // K = c 32..63

  unsigned ci[8];
#pragma unroll
  for (int r = 0; r < 8; ++r) ci[r] = code[i0 + r + 8 * hi];

  float mrun[8], lrun[8];
#pragma unroll
  for (int r = 0; r < 8; ++r) { mrun[r] = NEG; lrun[r] = 0.0f; }
  v8f o0 = {}, o1 = {}, o2 = {};

  for (int j0 = 0; j0 < NTOK; j0 += 32) {
    // ---- S = Q K^T for two 16-key tiles (K-dim 64 = 2 x 32) ----
    const _Float16* k0r = Kh + (size_t)(j0 + col) * DP;
    const _Float16* k1r = Kh + (size_t)(j0 + 16 + col) * DP;
    v8f s0v = {}, s1v = {};
    s0v = wmma_f16(qa0, *(const v16h*)(k0r + kb),      s0v);
    s0v = wmma_f16(qa1, *(const v16h*)(k0r + 32 + kb), s0v);
    s1v = wmma_f16(qa0, *(const v16h*)(k1r + kb),      s1v);
    s1v = wmma_f16(qa1, *(const v16h*)(k1r + 32 + kb), s1v);

    unsigned cj0 = code[j0 + col];
    unsigned cj1 = code[j0 + 16 + col];

    // ---- mask + online softmax (rows live across 16 lanes) ----
#pragma unroll
    for (int r = 0; r < 8; ++r) {
      int i  = i0 + r + 8 * hi;
      int jA = j0 + col, jB = j0 + 16 + col;
      float s0 = s0v[r] * scale;
      float s1 = s1v[r] * scale;
      bool keep0 = (i == jA) || ((ci[r] | cj0) & 0x8000u) || ((ci[r] & cj0 & 0x01FFu) != 0u);
      bool keep1 = (i == jB) || ((ci[r] | cj1) & 0x8000u) || ((ci[r] & cj1 & 0x01FFu) != 0u);
      if (!keep0) s0 = NEG;
      if (!keep1) s1 = NEG;

      float mloc = fmaxf(s0, s1);
#pragma unroll
      for (int off = 1; off < 16; off <<= 1)
        mloc = fmaxf(mloc, __shfl_xor(mloc, off, 32));
      float mnew  = fmaxf(mrun[r], mloc);
      float alpha = __expf(mrun[r] - mnew);
      float p0 = __expf(s0 - mnew);
      float p1 = __expf(s1 - mnew);
      float ps = p0 + p1;
#pragma unroll
      for (int off = 1; off < 16; off <<= 1)
        ps += __shfl_xor(ps, off, 32);
      lrun[r] = lrun[r] * alpha + ps;
      mrun[r] = mnew;
      o0[r] *= alpha; o1[r] *= alpha; o2[r] *= alpha;

      int m = r + 8 * hi;
      plds[h][m][col]      = (_Float16)p0;   // k = col
      plds[h][m][16 + col] = (_Float16)p1;   // k = 16+col
    }

#if __has_builtin(__builtin_amdgcn_s_wait_dscnt)
    __builtin_amdgcn_s_wait_dscnt(0);        // intra-wave DS RAW safety
#else
    __syncthreads();
#endif

    // ---- O += P(16x32) * V(32x48) : 3 N-tiles (c<48; c>=40 discarded) ----
    v16h pa = load_a_h(&plds[h][col][0], kh);
    o0 = wmma_f16(pa, *(const v16h*)(Vh + (size_t)(0  + col) * NTOK + j0 + kb), o0);
    o1 = wmma_f16(pa, *(const v16h*)(Vh + (size_t)(16 + col) * NTOK + j0 + kb), o1);
    o2 = wmma_f16(pa, *(const v16h*)(Vh + (size_t)(32 + col) * NTOK + j0 + kb), o2);
  }

  // ---- normalize + write per-head slice of O (f16, [N,320]) ----
#pragma unroll
  for (int r = 0; r < 8; ++r) {
    int i = i0 + r + 8 * hi;
    float inv = 1.0f / lrun[r];
    _Float16* orow = Of + (size_t)i * QD + h * DHD;
    orow[col]      = (_Float16)(o0[r] * inv);              // c = 0..15
    orow[16 + col] = (_Float16)(o1[r] * inv);              // c = 16..31
    if (col < 8) orow[32 + col] = (_Float16)(o2[r] * inv); // c = 32..39
  }
}

// ---------------- output projection: [N,320] x [320,320] + bias ----------------
__global__ void __launch_bounds__(640) k_out(const _Float16* Of, const _Float16* wot,
                                             const float* bo, float* out) {
  const int lane = threadIdx.x & 31;
  const int wid  = threadIdx.x >> 5;
  const int hi   = lane >> 4;
  const int col  = lane & 15;
  const int kh   = 8 * hi;
  const int kb   = 16 * hi;
  const int i0   = blockIdx.x * 16;
  const int n0   = wid * 16;

  v8f acc = {};
  const _Float16* orow = Of  + (size_t)(i0 + col) * QD;
  const _Float16* brow = wot + (size_t)(n0 + col) * QD;
#pragma unroll
  for (int k0 = 0; k0 < QD; k0 += 32) {
    v16h a = load_a_h(orow + k0, kh);
    v16h b = *(const v16h*)(brow + k0 + kb);
    acc = wmma_f16(a, b, acc);
  }
  float bias = bo[n0 + col];
#pragma unroll
  for (int r = 0; r < 8; ++r) {
    int i = i0 + r + 8 * hi;
    out[(size_t)i * QD + n0 + col] = acc[r] + bias;
  }
}

// ---------------- launch ----------------
extern "C" void kernel_launch(void* const* d_in, const int* in_sizes, int n_in,
                              void* d_out, int out_size, void* d_ws, size_t ws_size,
                              hipStream_t stream) {
  const float* x   = (const float*)d_in[0];  // [1,4128,320]
  const float* att = (const float*)d_in[1];  // [1,8,64,64]
  const float* Wq  = (const float*)d_in[2];
  const float* Wk  = (const float*)d_in[3];
  const float* Wv  = (const float*)d_in[4];
  const float* Wo  = (const float*)d_in[5];
  const float* bo  = (const float*)d_in[6];
  float* out = (float*)d_out;

  // workspace layout (~19 MB total), 256B aligned slabs
  const size_t QP_B = (size_t)NH * NTOK * DP * 2;   // 4,227,072
  const size_t WT_B = (size_t)QD * QD * 2;          //   204,800
  const size_t XF_B = (size_t)NTOK * QD * 2;        // 2,641,920
  char* ws = (char*)d_ws;
  size_t off = 0;
  auto take = [&](size_t bytes) -> void* {
    void* p = ws + off;
    off = (off + bytes + 255) & ~(size_t)255;
    return p;
  };
  _Float16* Qp  = (_Float16*)take(QP_B);
  _Float16* Kp  = (_Float16*)take(QP_B);
  _Float16* Vt  = (_Float16*)take(QP_B);
  _Float16* WqT = (_Float16*)take(WT_B);
  _Float16* WkT = (_Float16*)take(WT_B);
  _Float16* WvT = (_Float16*)take(WT_B);
  _Float16* WoT = (_Float16*)take(WT_B);
  _Float16* x16 = (_Float16*)take(XF_B);
  _Float16* Of  = (_Float16*)take(XF_B);
  unsigned short* code = (unsigned short*)take((size_t)NTOK * 2);
  (void)in_sizes; (void)n_in; (void)out_size; (void)ws_size;

  // 1) zero padded Q/K buffers (adjacent slabs -> single pass)
  long ndw = (long)(2 * QP_B / 4);
  k_zero<<<dim3((unsigned)((ndw + 255) / 256)), dim3(256), 0, stream>>>((uint32_t*)Qp, ndw);
  // 2) x -> f16, weight transposes, mask codes
  long nx = (long)NTOK * QD;
  k_x16<<<dim3((unsigned)((nx + 255) / 256)), dim3(256), 0, stream>>>(x, x16, nx);
  k_wt<<<dim3((QD * QD + 255) / 256), dim3(256), 0, stream>>>(Wq, Wk, Wv, Wo, WqT, WkT, WvT, WoT);
  k_code<<<dim3((NTOK + 255) / 256), dim3(256), 0, stream>>>(att, code);
  // 3) QKV projection
  k_qkv<<<dim3(NTOK / 16), dim3(640), 0, stream>>>(x16, WqT, WkT, WvT, Qp, Kp, Vt);
  // 4) masked flash attention
  k_attn<<<dim3(NTOK / 16), dim3(256), 0, stream>>>(Qp, Kp, Vt, code, Of);
  // 5) output projection + bias
  k_out<<<dim3(NTOK / 16), dim3(640), 0, stream>>>(Of, WoT, bo, out);
}
